// LinearRNN_51883204935738
// MI455X (gfx1250) — compile-verified, matrix operand order checked by reading, not run on hardware
//
#include <hip/hip_runtime.h>

typedef __bf16 bf16;
typedef __attribute__((ext_vector_type(16))) bf16 v16bf;
typedef __attribute__((ext_vector_type(8)))  float v8f;
typedef int v4i __attribute__((vector_size(16)));

#define DIMN 512
#define TLEN 4096
#define NCHUNK 128        // chunks over T
#define CLEN 32           // steps per chunk
#define LDSP 516          // padded LDS row stride (floats)
#define BATCH_ELEMS 4096  // 8 batches * 512 dims

#if defined(__has_builtin)
#if __has_builtin(__builtin_amdgcn_global_load_async_to_lds_b128) && \
    __has_builtin(__builtin_amdgcn_s_wait_asynccnt)
#define HAVE_ASYNC 1
#endif
#endif
#ifndef HAVE_ASYNC
#define HAVE_ASYNC 0
#endif

#if HAVE_ASYNC
typedef __attribute__((address_space(1))) v4i as1_v4i;
typedef __attribute__((address_space(3))) v4i as3_v4i;
#endif

// ---- WMMA fragment helpers (CDNA5 16-bit A/B layout) ------------------------
__device__ __forceinline__ v16bf frag_from_f32(const float* rowptr, int k0, int half) {
  v16bf a;
  const float* p0 = rowptr + k0 + half * 8;
  const float* p1 = p0 + 16;
#pragma unroll
  for (int e = 0; e < 8; ++e) { a[e] = (bf16)p0[e]; a[e + 8] = (bf16)p1[e]; }
  return a;
}

__device__ __forceinline__ v16bf frag_from_bf16(const bf16* rowptr, int k0, int half) {
  v16bf a;
  const bf16* p0 = rowptr + k0 + half * 8;
  const bf16* p1 = p0 + 16;
#pragma unroll
  for (int e = 0; e < 8; ++e) { a[e] = p0[e]; a[e + 8] = p1[e]; }
  return a;
}

__device__ __forceinline__ v8f wmma_bf16(v16bf a, v16bf b, v8f c) {
  return __builtin_amdgcn_wmma_f32_16x16x32_bf16(false, a, false, b, (short)0, c,
                                                 false, false);
}

// ---- Async staging: copy one 16x64-f32 tile (this wave's Wx slice) to LDS ---
// Per wave: 8 b128 ops; op i covers rows {2i, 2i+1}; lanes 0-15 row 2i,
// lanes 16-31 row 2i+1; each lane moves 16 bytes.
__device__ __forceinline__ void stage_wx(const float* __restrict__ out, int c0,
                                         int s, int wave, int lane,
                                         float* __restrict__ stage /*16x64*/) {
  const int rsel = lane >> 4;
  const int cseg = (lane & 15) * 4;
#pragma unroll
  for (int i = 0; i < 8; ++i) {
    int m = 2 * i + rsel;
    int chunk = c0 + (m >> 3);
    int bb = m & 7;
    int t = chunk * CLEN + s;
    const float* g = out + ((size_t)bb * TLEN + t) * DIMN + wave * 64 + cseg;
    float* l = stage + m * 64 + cseg;
#if HAVE_ASYNC
    __builtin_amdgcn_global_load_async_to_lds_b128((as1_v4i*)g, (as3_v4i*)l, 0, 0);
#else
    *(float4*)l = *(const float4*)g;
#endif
  }
}

__device__ __forceinline__ void stage_wait_prev() {
#if HAVE_ASYNC
  __builtin_amdgcn_s_wait_asynccnt(8);   // prev step's 8 ops done (in-order)
#endif
}
__device__ __forceinline__ void stage_wait_all() {
#if HAVE_ASYNC
  __builtin_amdgcn_s_wait_asynccnt(0);
#endif
}

// ---- Kernel 1: convert fp32 matrix to bf16 (row-major, optional transpose) --
__global__ void k_cvt(const float* __restrict__ S, bf16* __restrict__ Dr,
                      bf16* __restrict__ Dt) {
  int idx = blockIdx.x * blockDim.x + threadIdx.x;   // 512*512
  int r = idx >> 9, c = idx & 511;
  float v = S[idx];
  Dr[idx] = (bf16)v;
  if (Dt) Dt[c * DIMN + r] = (bf16)v;
}

// ---- Kernel 2: Wx = x @ W_w^T + (W_b + U_b) into d_out ----------------------
__global__ void k_wx(const float* __restrict__ x, const bf16* __restrict__ Wwbf,
                     const float* __restrict__ Wb, const float* __restrict__ Ub,
                     float* __restrict__ out) {
  const int lane = threadIdx.x & 31;
  const int wave = threadIdx.x >> 5;
  const int lr = lane & 15, half = lane >> 4;
  const int m0 = (blockIdx.x * 8 + wave) * 16;
  const int n0 = blockIdx.y * 128;
  const float* arow = x + (size_t)(m0 + lr) * DIMN;
  v8f acc[8] = {};
  for (int k0 = 0; k0 < DIMN; k0 += 32) {
    v16bf a = frag_from_f32(arow, k0, half);
#pragma unroll
    for (int nt = 0; nt < 8; ++nt) {
      v16bf b = frag_from_bf16(Wwbf + (size_t)(n0 + nt * 16 + lr) * DIMN, k0, half);
      acc[nt] = wmma_bf16(a, b, acc[nt]);
    }
  }
#pragma unroll
  for (int nt = 0; nt < 8; ++nt) {
    int col = n0 + nt * 16 + lr;
    float bias = Wb[col] + Ub[col];
#pragma unroll
    for (int r = 0; r < 8; ++r) {
      int orow = m0 + r + half * 8;
      out[(size_t)orow * DIMN + col] = acc[nt][r] + bias;
    }
  }
}

// ---- Kernel 3: C = A*A (row-major + transposed copies in/out) ---------------
__global__ void k_square(const bf16* __restrict__ Ar, const bf16* __restrict__ At,
                         bf16* __restrict__ Cr, bf16* __restrict__ Ct) {
  const int lane = threadIdx.x & 31;
  const int wave = threadIdx.x >> 5;
  const int lr = lane & 15, half = lane >> 4;
  const int m0 = (blockIdx.x * 8 + wave) * 16;
  const int n0 = blockIdx.y * 128;
  const bf16* arow = Ar + (size_t)(m0 + lr) * DIMN;
  v8f acc[8] = {};
  for (int k0 = 0; k0 < DIMN; k0 += 32) {
    v16bf a = frag_from_bf16(arow, k0, half);
#pragma unroll
    for (int nt = 0; nt < 8; ++nt) {
      v16bf b = frag_from_bf16(At + (size_t)(n0 + nt * 16 + lr) * DIMN, k0, half);
      acc[nt] = wmma_bf16(a, b, acc[nt]);
    }
  }
#pragma unroll
  for (int nt = 0; nt < 8; ++nt) {
    int ocol = n0 + nt * 16 + lr;
#pragma unroll
    for (int r = 0; r < 8; ++r) {
      int orow = m0 + r + half * 8;
      bf16 v = (bf16)acc[nt][r];
      Cr[(size_t)orow * DIMN + ocol] = v;
      Ct[(size_t)ocol * DIMN + orow] = v;
    }
  }
}

// ---- Kernel 4: local chunk scan (zero carry-in), in place in d_out ----------
// Block handles chunks (2bx, 2bx+1) packed as WMMA rows 0-7 / 8-15.
// Wx_t slabs are double-buffer prefetched into LDS with async-to-LDS ops.
__global__ void k_scan_local(const bf16* __restrict__ Ubf, float* __restrict__ out) {
  __shared__ float Hs[16 * LDSP];
  __shared__ float Wstage[2][8][16 * 64];
  const int lane = threadIdx.x & 31;
  const int wave = threadIdx.x >> 5;
  const int lr = lane & 15, half = lane >> 4;
  const int c0 = blockIdx.x * 2;
  for (int i = threadIdx.x; i < 16 * LDSP; i += blockDim.x) Hs[i] = 0.0f;
  __syncthreads();
  const int ncol0 = wave * 64;
  stage_wx(out, c0, 0, wave, lane, Wstage[0][wave]);
  for (int s = 0; s < CLEN; ++s) {
    const int buf = s & 1;
    v16bf afrag[16];
#pragma unroll
    for (int kt = 0; kt < 16; ++kt)
      afrag[kt] = frag_from_f32(&Hs[lr * LDSP], kt * 32, half);
    __syncthreads();
    if (s + 1 < CLEN) stage_wx(out, c0, s + 1, wave, lane, Wstage[buf ^ 1][wave]);
    v8f acc[4] = {};
#pragma unroll
    for (int nt = 0; nt < 4; ++nt) {
      const bf16* brow = Ubf + (size_t)(ncol0 + nt * 16 + lr) * DIMN;
#pragma unroll
      for (int kt = 0; kt < 16; ++kt) {
        v16bf b = frag_from_bf16(brow, kt * 32, half);
        acc[nt] = wmma_bf16(afrag[kt], b, acc[nt]);
      }
    }
    if (s + 1 < CLEN) stage_wait_prev(); else stage_wait_all();
    const float* wxs = Wstage[buf][wave];
#pragma unroll
    for (int nt = 0; nt < 4; ++nt) {
      int col = ncol0 + nt * 16 + lr;
#pragma unroll
      for (int r = 0; r < 8; ++r) {
        int m = r + half * 8;
        int chunk = c0 + (m >> 3);
        int bb = m & 7;
        int t = chunk * CLEN + s;
        size_t gi = ((size_t)bb * TLEN + t) * DIMN + col;
        float h = acc[nt][r] + wxs[m * 64 + (col - ncol0)];
        out[gi] = h;
        Hs[m * LDSP + col] = h;
      }
    }
    __syncthreads();
  }
}

// ---- Kernel 5: boundary scan E_j = localEnd_j + E_{j-1} * V^CLEN ------------
__global__ void k_boundary(const bf16* __restrict__ Gbf, const float* __restrict__ outc,
                           float* __restrict__ Carry) {
  __shared__ float Es[16 * LDSP];
  const int lane = threadIdx.x & 31;
  const int wave = threadIdx.x >> 5;
  const int lr = lane & 15, half = lane >> 4;
  for (int i = threadIdx.x; i < 16 * LDSP; i += blockDim.x) Es[i] = 0.0f;
  __syncthreads();
  const int ncol0 = wave * 64;
  for (int j = 0; j < NCHUNK; ++j) {
    for (int i = threadIdx.x; i < BATCH_ELEMS; i += blockDim.x) {
      int r = i >> 9, c = i & 511;
      Carry[(size_t)j * BATCH_ELEMS + i] = Es[r * LDSP + c];
    }
    v16bf afrag[16];
#pragma unroll
    for (int kt = 0; kt < 16; ++kt)
      afrag[kt] = frag_from_f32(&Es[lr * LDSP], kt * 32, half);
    __syncthreads();
    v8f acc[4] = {};
#pragma unroll
    for (int nt = 0; nt < 4; ++nt) {
      const bf16* brow = Gbf + (size_t)(ncol0 + nt * 16 + lr) * DIMN;
#pragma unroll
      for (int kt = 0; kt < 16; ++kt) {
        v16bf b = frag_from_bf16(brow, kt * 32, half);
        acc[nt] = wmma_bf16(afrag[kt], b, acc[nt]);
      }
    }
#pragma unroll
    for (int nt = 0; nt < 4; ++nt) {
      int col = ncol0 + nt * 16 + lr;
#pragma unroll
      for (int r = 0; r < 8; ++r) {
        int m = r + half * 8;
        if (m < 8) {
          int t = j * CLEN + (CLEN - 1);
          size_t gi = ((size_t)m * TLEN + t) * DIMN + col;
          Es[m * LDSP + col] = acc[nt][r] + outc[gi];
        }
      }
    }
    __syncthreads();
  }
}

// ---- Kernel 6: correction pass: corr <- corr * V, out += corr ---------------
__global__ void k_correct(const bf16* __restrict__ Ubf, const float* __restrict__ Carry,
                          float* __restrict__ out) {
  __shared__ float Hs[16 * LDSP];
  __shared__ float Wstage[2][8][16 * 64];
  const int lane = threadIdx.x & 31;
  const int wave = threadIdx.x >> 5;
  const int lr = lane & 15, half = lane >> 4;
  const int c0 = blockIdx.x * 2;
  for (int i = threadIdx.x; i < BATCH_ELEMS; i += blockDim.x) {
    int r = i >> 9, c = i & 511;
    Hs[r * LDSP + c]       = Carry[(size_t)c0 * BATCH_ELEMS + i];
    Hs[(r + 8) * LDSP + c] = Carry[(size_t)(c0 + 1) * BATCH_ELEMS + i];
  }
  __syncthreads();
  const int ncol0 = wave * 64;
  stage_wx(out, c0, 0, wave, lane, Wstage[0][wave]);
  for (int s = 0; s < CLEN; ++s) {
    const int buf = s & 1;
    v16bf afrag[16];
#pragma unroll
    for (int kt = 0; kt < 16; ++kt)
      afrag[kt] = frag_from_f32(&Hs[lr * LDSP], kt * 32, half);
    __syncthreads();
    if (s + 1 < CLEN) stage_wx(out, c0, s + 1, wave, lane, Wstage[buf ^ 1][wave]);
    v8f acc[4] = {};
#pragma unroll
    for (int nt = 0; nt < 4; ++nt) {
      const bf16* brow = Ubf + (size_t)(ncol0 + nt * 16 + lr) * DIMN;
#pragma unroll
      for (int kt = 0; kt < 16; ++kt) {
        v16bf b = frag_from_bf16(brow, kt * 32, half);
        acc[nt] = wmma_bf16(afrag[kt], b, acc[nt]);
      }
    }
    if (s + 1 < CLEN) stage_wait_prev(); else stage_wait_all();
    const float* wxs = Wstage[buf][wave];
#pragma unroll
    for (int nt = 0; nt < 4; ++nt) {
      int col = ncol0 + nt * 16 + lr;
#pragma unroll
      for (int r = 0; r < 8; ++r) {
        int m = r + half * 8;
        int chunk = c0 + (m >> 3);
        int bb = m & 7;
        int t = chunk * CLEN + s;
        size_t gi = ((size_t)bb * TLEN + t) * DIMN + col;
        float corr = acc[nt][r];
        out[gi] = corr + wxs[m * 64 + (col - ncol0)];
        Hs[m * LDSP + col] = corr;
      }
    }
    __syncthreads();
  }
}

extern "C" void kernel_launch(void* const* d_in, const int* in_sizes, int n_in,
                              void* d_out, int out_size, void* d_ws, size_t ws_size,
                              hipStream_t stream) {
  const float* x  = (const float*)d_in[0];
  const float* Ww = (const float*)d_in[1];
  const float* Wb = (const float*)d_in[2];
  const float* Uw = (const float*)d_in[3];
  const float* Ub = (const float*)d_in[4];
  float* out = (float*)d_out;
  char* ws = (char*)d_ws;

  bf16*  Ubf   = (bf16*)(ws);                     // 512 KB
  bf16*  UbfT  = (bf16*)(ws + 512ull * 1024);     // 512 KB
  bf16*  PA    = (bf16*)(ws + 1024ull * 1024);
  bf16*  PAT   = (bf16*)(ws + 1536ull * 1024);
  bf16*  PB    = (bf16*)(ws + 2048ull * 1024);
  bf16*  PBT   = (bf16*)(ws + 2560ull * 1024);
  float* Carry = (float*)(ws + 3072ull * 1024);   // 128 * 4096 f32 = 2 MB
  bf16*  Wwbf  = (bf16*)(ws + 5120ull * 1024);    // 512 KB

  k_cvt<<<1024, 256, 0, stream>>>(Uw, Ubf, UbfT);
  k_cvt<<<1024, 256, 0, stream>>>(Ww, Wwbf, (bf16*)nullptr);
  k_wx<<<dim3(256, 4), 256, 0, stream>>>(x, Wwbf, Wb, Ub, out);
  // U^2, U^4, U^8, U^16, U^32 by repeated squaring; U^32 row-major ends in PA
  k_square<<<dim3(4, 4), 256, 0, stream>>>(Ubf, UbfT, PA, PAT);
  k_square<<<dim3(4, 4), 256, 0, stream>>>(PA, PAT, PB, PBT);
  k_square<<<dim3(4, 4), 256, 0, stream>>>(PB, PBT, PA, PAT);
  k_square<<<dim3(4, 4), 256, 0, stream>>>(PA, PAT, PB, PBT);
  k_square<<<dim3(4, 4), 256, 0, stream>>>(PB, PBT, PA, PAT);
  k_scan_local<<<NCHUNK / 2, 256, 0, stream>>>(Ubf, out);
  k_boundary<<<1, 256, 0, stream>>>(PA, out, Carry);
  k_correct<<<NCHUNK / 2, 256, 0, stream>>>(Ubf, Carry, out);
}
